// GCNEncoder_10823317586229
// MI455X (gfx1250) — matrix operand order repaired and checked by reading, not compiled, not run on hardware
//
#include <hip/hip_runtime.h>
#include <hip/hip_bf16.h>

typedef __attribute__((ext_vector_type(16))) __bf16 v16bf;
typedef __attribute__((ext_vector_type(8)))  float  v8f;

// ---------------- degree / dinv / edge norm ----------------
__global__ void deg_init(float* deg, int Nn) {
    int i = blockIdx.x * blockDim.x + threadIdx.x;
    if (i < Nn) deg[i] = 1.0f;   // self loop
}
__global__ void deg_count(const int* __restrict__ dst, float* __restrict__ deg, int Ee) {
    int i = blockIdx.x * blockDim.x + threadIdx.x;
    if (i < Ee) atomicAdd(&deg[dst[i]], 1.0f);
}
__global__ void deg_rsqrt(float* deg, int Nn) {
    int i = blockIdx.x * blockDim.x + threadIdx.x;
    if (i < Nn) deg[i] = rsqrtf(deg[i]);
}
__global__ void edge_norm_k(const int* __restrict__ src, const int* __restrict__ dst,
                            const float* __restrict__ dinv, float* __restrict__ norm, int Ee) {
    int i = blockIdx.x * blockDim.x + threadIdx.x;
    if (i < Ee) norm[i] = dinv[src[i]] * dinv[dst[i]];
}

// ---------------- dense GEMM via WMMA (bf16 in, f32 acc) ----------------
// O[Nrows x FOUT] = X[Nrows x 128] @ W[128 x FOUT].
// One wave owns one 16-wide column tile and MT row tiles; the full K=128
// B-operand (4 x v16bf = 32 VGPRs) is loaded once (immediate-offset clause
// loads thanks to compile-time FOUT) and reused across all MT row tiles.
template <int FOUT, int MT>
__global__ void gemm128_wmma(const float* __restrict__ X, const float* __restrict__ W,
                             float* __restrict__ O, int Nrows) {
    const int lane = threadIdx.x & 31;
    const int wave = (blockIdx.x * blockDim.x + threadIdx.x) >> 5;
    const int tiles_n = FOUT / 16;
    const int mg = wave / tiles_n;         // row-tile group
    const int tn = wave % tiles_n;
    const int tm0 = mg * MT;
    if (tm0 * 16 >= Nrows) return;         // wave-uniform
    const int half = lane >> 4;            // 0: lanes 0-15, 1: lanes 16-31
    const int l15  = lane & 15;
    const int col  = tn * 16 + l15;

    // B: 32x16 bf16 tiles; lanes 0-15 K=k0..k0+15, lanes 16-31 K=k0+16..k0+31
    v16bf b[4];
#pragma unroll
    for (int ks = 0; ks < 4; ++ks) {
        const float* __restrict__ bp = W + (size_t)(ks * 32 + half * 16) * FOUT + col;
#pragma unroll
        for (int j = 0; j < 16; ++j) b[ks][j] = (__bf16)bp[j * FOUT];  // imm offsets
    }

#pragma unroll
    for (int t = 0; t < MT; ++t) {
        const int tm = tm0 + t;
        if (tm * 16 >= Nrows) break;       // wave-uniform
        int arowi = tm * 16 + l15;
        if (arowi >= Nrows) arowi = Nrows - 1;   // clamp loads, keep EXEC all-ones
        const float* __restrict__ arow = X + (size_t)arowi * 128;

        v8f c = {};
#pragma unroll
        for (int ks = 0; ks < 4; ++ks) {
            const int k0 = ks * 32;
            // A: 16x32 bf16 tile; lanes 0-15: K 0..15, lanes 16-31: K 16..31
            const float4* alo = (const float4*)(arow + k0 + half * 16);
            const float4* ahi = (const float4*)(arow + k0 + half * 16 + 8);
            float4 p0 = alo[0], p1 = alo[1];
            float4 q0 = ahi[0], q1 = ahi[1];
            v16bf a;
            a[0]  = (__bf16)p0.x; a[1]  = (__bf16)p0.y; a[2]  = (__bf16)p0.z; a[3]  = (__bf16)p0.w;
            a[4]  = (__bf16)p1.x; a[5]  = (__bf16)p1.y; a[6]  = (__bf16)p1.z; a[7]  = (__bf16)p1.w;
            a[8]  = (__bf16)q0.x; a[9]  = (__bf16)q0.y; a[10] = (__bf16)q0.z; a[11] = (__bf16)q0.w;
            a[12] = (__bf16)q1.x; a[13] = (__bf16)q1.y; a[14] = (__bf16)q1.z; a[15] = (__bf16)q1.w;

            c = __builtin_amdgcn_wmma_f32_16x16x32_bf16(false, a, false, b[ks],
                                                        (short)0, c, false, false);
        }

        // D/C layout: VGPR r -> row (tm*16 + r + half*8), column tn*16 + l15
#pragma unroll
        for (int r = 0; r < 8; ++r) {
            int rr = tm * 16 + half * 8 + r;
            if (rr < Nrows) O[(size_t)rr * FOUT + col] = c[r];
        }
    }
}

// ---------------- aggregation ----------------
// agg = h * dinv^2 (self loop) + bias     (also clears poison in agg/d_out)
__global__ void init_agg(const float* __restrict__ h, const float* __restrict__ dinv,
                         const float* __restrict__ bias, float* __restrict__ agg,
                         int Nn, int F) {
    size_t i = (size_t)blockIdx.x * blockDim.x + threadIdx.x;
    size_t total = (size_t)Nn * (size_t)F;
    if (i >= total) return;
    int row = (int)(i / F);
    int f   = (int)(i % F);
    float di = dinv[row];
    agg[i] = h[i] * di * di + bias[f];
}

// agg[dst] += h[src] * norm[e]; (F/4) threads per edge, float4 gathers + f32 atomics
__global__ void scatter_edges(const float* __restrict__ h, const int* __restrict__ src,
                              const int* __restrict__ dst, const float* __restrict__ norm,
                              float* __restrict__ agg, int Ee, int F) {
    const int chunks = F >> 2;
    long long tid = (long long)blockIdx.x * blockDim.x + threadIdx.x;
    int e = (int)(tid / chunks);
    int c = (int)(tid % chunks);
    if (e >= Ee) return;
    int s = src[e], d = dst[e];
    float w = norm[e];
    float4 v = *(const float4*)(h + (size_t)s * F + (size_t)c * 4);
    float* o = agg + (size_t)d * F + (size_t)c * 4;
    atomicAdd(o + 0, v.x * w);
    atomicAdd(o + 1, v.y * w);
    atomicAdd(o + 2, v.z * w);
    atomicAdd(o + 3, v.w * w);
}

// ---------------- batchnorm (+ReLU), in place ----------------
__global__ void bn_zero(float* sums, int n) {
    int i = blockIdx.x * blockDim.x + threadIdx.x;
    if (i < n) sums[i] = 0.0f;
}
__global__ void bn_stats(const float* __restrict__ x, float* __restrict__ sums,
                         float* __restrict__ sumsq, int Nn, int F) {
    int f  = threadIdx.x;               // blockDim.x == F
    int r0 = blockIdx.x * 256;
    int r1 = r0 + 256; if (r1 > Nn) r1 = Nn;
    float s = 0.0f, s2 = 0.0f;
    for (int r = r0; r < r1; ++r) {
        float v = x[(size_t)r * F + f];
        s += v; s2 += v * v;
    }
    atomicAdd(&sums[f], s);
    atomicAdd(&sumsq[f], s2);
}
__global__ void bn_finalize(const float* __restrict__ sums, const float* __restrict__ sumsq,
                            const float* __restrict__ g, const float* __restrict__ be,
                            float* __restrict__ scale, float* __restrict__ shift,
                            int F, float invN) {
    int f = blockIdx.x * blockDim.x + threadIdx.x;
    if (f >= F) return;
    float mu  = sums[f] * invN;
    float var = sumsq[f] * invN - mu * mu;
    float rs  = rsqrtf(var + 1e-5f);
    float sc  = rs * g[f];
    scale[f] = sc;
    shift[f] = be[f] - mu * sc;
}
__global__ void bn_apply_relu(float* __restrict__ x, const float* __restrict__ scale,
                              const float* __restrict__ shift, int Nn, int F) {
    size_t i = (size_t)blockIdx.x * blockDim.x + threadIdx.x;
    size_t total = (size_t)Nn * (size_t)F;
    if (i >= total) return;
    int f = (int)(i % F);
    float v = fmaf(x[i], scale[f], shift[f]);
    x[i] = v > 0.0f ? v : 0.0f;
}

static inline int cdiv(long long a, long long b) { return (int)((a + b - 1) / b); }

extern "C" void kernel_launch(void* const* d_in, const int* in_sizes, int n_in,
                              void* d_out, int out_size, void* d_ws, size_t ws_size,
                              hipStream_t stream) {
    const float* x  = (const float*)d_in[0];
    const int*   ei = (const int*)d_in[1];
    const float* W1 = (const float*)d_in[2];  const float* b1 = (const float*)d_in[3];
    const float* W2 = (const float*)d_in[4];  const float* b2 = (const float*)d_in[5];
    const float* W3 = (const float*)d_in[6];  const float* b3 = (const float*)d_in[7];
    const float* W4 = (const float*)d_in[8];  const float* b4 = (const float*)d_in[9];
    const float* g1 = (const float*)d_in[10]; const float* be1 = (const float*)d_in[11];
    const float* g2 = (const float*)d_in[12]; const float* be2 = (const float*)d_in[13];
    const float* g3 = (const float*)d_in[14]; const float* be3 = (const float*)d_in[15];

    const int IN = 128, H = 128, OUT = 64;
    const int N = in_sizes[0] / IN;
    const int E = in_sizes[1] / 2;
    const int* src = ei;
    const int* dst = ei + E;

    // workspace layout (floats)
    float* f     = (float*)d_ws;
    float* dinv  = f;
    float* norm  = dinv + (((size_t)N + 63) & ~(size_t)63);
    float* h     = norm + (((size_t)E + 63) & ~(size_t)63);
    float* agg   = h + (size_t)N * H;
    float* sums  = agg + (size_t)N * H;
    float* sumsq = sums + H;
    float* scale = sumsq + H;
    float* shift = scale + H;

    const int TB = 256;
    const int MT = 8;                        // row tiles per wave in the GEMM
    const float invN = 1.0f / (float)N;

    // degree (with self loop) -> dinv = rsqrt(deg) -> per-edge norm
    deg_init   <<<cdiv(N, TB), TB, 0, stream>>>(dinv, N);
    deg_count  <<<cdiv(E, TB), TB, 0, stream>>>(dst, dinv, E);
    deg_rsqrt  <<<cdiv(N, TB), TB, 0, stream>>>(dinv, N);
    edge_norm_k<<<cdiv(E, TB), TB, 0, stream>>>(src, dst, dinv, norm, E);

    const float* Ws[3]  = {W1, W2, W3};
    const float* bs[3]  = {b1, b2, b3};
    const float* gs[3]  = {g1, g2, g3};
    const float* bes[3] = {be1, be2, be3};

    const long long waves128 = (long long)cdiv(N, 16 * MT) * MT /*row groups spread*/;
    // waves = row_groups * tiles_n; compute explicitly:
    const long long rg = cdiv(N, 16 * MT);

    const float* cur = x;
    for (int l = 0; l < 3; ++l) {
        (void)waves128;
        long long waves = rg * (H / 16);
        gemm128_wmma<128, 8><<<cdiv(waves * 32, TB), TB, 0, stream>>>(cur, Ws[l], h, N);
        init_agg     <<<cdiv((long long)N * H, TB), TB, 0, stream>>>(h, dinv, bs[l], agg, N, H);
        scatter_edges<<<cdiv((long long)E * (H / 4), TB), TB, 0, stream>>>(h, src, dst, norm, agg, E, H);
        bn_zero      <<<1, 2 * H, 0, stream>>>(sums, 2 * H);
        bn_stats     <<<cdiv(N, 256), H, 0, stream>>>(agg, sums, sumsq, N, H);
        bn_finalize  <<<1, H, 0, stream>>>(sums, sumsq, gs[l], bes[l], scale, shift, H, invN);
        bn_apply_relu<<<cdiv((long long)N * H, TB), TB, 0, stream>>>(agg, scale, shift, N, H);
        cur = agg;
    }

    // layer 4: GEMM(128->64) + aggregation straight into d_out (no BN/ReLU)
    float* out = (float*)d_out;
    long long waves4 = rg * (OUT / 16);
    gemm128_wmma<64, 8><<<cdiv(waves4 * 32, TB), TB, 0, stream>>>(cur, W4, h, N);
    init_agg     <<<cdiv((long long)N * OUT, TB), TB, 0, stream>>>(h, dinv, b4, out, N, OUT);
    scatter_edges<<<cdiv((long long)E * (OUT / 4), TB), TB, 0, stream>>>(h, src, dst, norm, out, E, OUT);
}
